// SemanticDecoupling_47081431499172
// MI455X (gfx1250) — matrix operand order repaired and checked by reading, not compiled, not run on hardware
//
#include <hip/hip_runtime.h>
#include <math.h>

typedef __attribute__((ext_vector_type(16))) _Float16 v16h;
typedef __attribute__((ext_vector_type(8)))  _Float16 v8h;
typedef __attribute__((ext_vector_type(4)))  _Float16 v4h;
typedef __attribute__((ext_vector_type(8)))  float    v8f;
typedef __attribute__((ext_vector_type(4)))  float    v4f;

namespace {
constexpr int kBS = 2, kS = 196, kD = 512, kC = 1006;
constexpr int kEHalf   = 256;                    // e-columns per block
constexpr int kCT      = 16;                     // c-rows per block
constexpr int kThreads = 256;                    // 8 wave32
constexpr int kCTiles  = (kC + kCT - 1) / kCT;   // 63
// W half-slice + double-buffered A tile
constexpr size_t kLdsHalfs = (size_t)kEHalf * kD + 2u * kCT * kD;  // 294912 B
}

// Hardware tanh: gfx1250 has V_TANH_F32 (trans op, co-executes with XDL WMMA).
__device__ __forceinline__ float hw_tanhf(float x) {
#if __has_builtin(__builtin_amdgcn_tanhf)
  return __builtin_amdgcn_tanhf(x);
#else
  float y;
  asm("v_tanh_f32 %0, %1\n\tv_nop" : "=v"(y) : "v"(x));
  return y;
#endif
}

// out[b,c,e] = sum_s img[b,s,e] * softmax_s( sum_k tanh(img[b,s,k]*word[c,k]) * W[e,k] + bias[e] )
__global__ __launch_bounds__(kThreads) void semdec_wmma_kernel(
    const float* __restrict__ img,   // (BS,S,D)
    const float* __restrict__ word,  // (C,D)
    const float* __restrict__ fw,    // (D,D) row-major [e][k]
    const float* __restrict__ fb,    // (D)
    float* __restrict__ out)         // (BS,C,D)
{
  extern __shared__ _Float16 smem[];
  _Float16* __restrict__ w_lds = smem;                       // [kEHalf][kD]
  _Float16* __restrict__ a_lds0 = smem + (size_t)kEHalf*kD;  // [kCT][kD] ping
  _Float16* __restrict__ a_lds1 = a_lds0 + (size_t)kCT*kD;   // [kCT][kD] pong

  const int tid  = threadIdx.x;
  const int wave = tid >> 5;
  const int lane = tid & 31;
  const int lhal = lane >> 4;   // lane half (0/1)
  const int l16  = lane & 15;

  const int cbase = blockIdx.x * kCT;
  const int ebase = blockIdx.y * kEHalf;
  const int b     = blockIdx.z;
  const float* __restrict__ imgb = img + (size_t)b * kS * kD;

  // ---- stage fc3_w e-slice into LDS as f16 (once per block) ----
  #pragma unroll 4
  for (int it = 0; it < (kEHalf * kD / 4) / kThreads; ++it) {  // 128 iters
    int idx4 = tid + it * kThreads;
    int el = idx4 >> 7;              // kD/4 == 128 float4 per row
    int k4 = (idx4 & 127) << 2;
    v4f wv = *(const v4f*)(fw + (size_t)(ebase + el) * kD + k4);
    v4h hv;
    hv.x = (_Float16)wv.x; hv.y = (_Float16)wv.y;
    hv.z = (_Float16)wv.z; hv.w = (_Float16)wv.w;
    *(v4h*)(w_lds + el * kD + k4) = hv;
  }

  // Per-thread A-build addressing (fixed across s).
  const int a_c  = tid >> 7;                 // two quad-columns per c-row pass
  const int a_k4 = (tid & 127) << 2;
  // builds rows a_c, a_c+2, ..., a_c+14 (8 iters of stride-2 rows)

  // Build tanh A-tile for spatial position s into `dst`.
  auto build_tile = [&](int s, _Float16* __restrict__ dst) {
    const float* __restrict__ imgrow = imgb + (size_t)s * kD;
    #pragma unroll
    for (int it = 0; it < 8; ++it) {
      int c  = a_c + it * 2;
      int cg = cbase + c; cg = (cg < kC) ? cg : (kC - 1);  // pad last tile
      v4f iv = *(const v4f*)(imgrow + a_k4);
      v4f wv = *(const v4f*)(word + (size_t)cg * kD + a_k4);
      v4h hv;
      hv.x = (_Float16)hw_tanhf(iv.x * wv.x);
      hv.y = (_Float16)hw_tanhf(iv.y * wv.y);
      hv.z = (_Float16)hw_tanhf(iv.z * wv.z);
      hv.w = (_Float16)hw_tanhf(iv.w * wv.w);
      *(v4h*)(dst + c * kD + a_k4) = hv;
    }
  };

  // Each wave owns two 16-wide e-tiles: columns e0 and e0+16 (N = lane&15).
  const int e0 = ebase + wave * 32 + l16;
  const float bias0 = fb[e0], bias1 = fb[e0 + 16];

  // Online-softmax state per (c,e) element: D-matrix slot (vgpr v, this lane).
  v8f   n0 = {}, n1 = {};
  float m0[8], m1[8], l0[8], l1[8];
  #pragma unroll
  for (int v = 0; v < 8; ++v) { m0[v] = m1[v] = -INFINITY; l0[v] = l1[v] = 0.f; }

  // Prologue: build tile for s=0 into ping buffer.
  build_tile(0, a_lds0);

  for (int s = 0; s < kS; ++s) {
    // Single barrier per iteration: guarantees (a) buf[s&1] fully built,
    // (b) everyone consumed their reads of buf[(s+1)&1] from iteration s-1.
    __syncthreads();
    const _Float16* __restrict__ cur = (s & 1) ? a_lds1 : a_lds0;
    _Float16* __restrict__ nxt       = (s & 1) ? a_lds0 : a_lds1;

    // feat tile = A(16x512) x B(512x16) via 16 chained WMMAs per e-tile
    v8f acc0 = {}, acc1 = {};
    #pragma unroll
    for (int kc = 0; kc < kD / 32; ++kc) {
      // A k order used here (contiguous, same as B): lanes 0-15 -> K 0..15, lanes 16-31 -> K 16..31
      const _Float16* ap = cur + l16 * kD + kc * 32 + lhal * 16;
      v8h alo = *(const v8h*)ap;          // ds_load_b128
      v8h ahi = *(const v8h*)(ap + 8);
      v16h av = __builtin_shufflevector(alo, ahi,
                  0,1,2,3,4,5,6,7,8,9,10,11,12,13,14,15);
      // B k order used here (contiguous, same as A): N = lane&15; lanes 0-15 -> K 0..15, lanes 16-31 -> K 16..31
      const _Float16* bp = w_lds + (wave * 32 + l16) * kD + kc * 32 + lhal * 16;
      v16h bv0 = *(const v16h*)bp;            // 2x ds_load_b128
      v16h bv1 = *(const v16h*)(bp + 16 * kD);
      acc0 = __builtin_amdgcn_wmma_f32_16x16x32_f16(false, av, false, bv0,
                                                    (short)0, acc0, false, false);
      acc1 = __builtin_amdgcn_wmma_f32_16x16x32_f16(false, av, false, bv1,
                                                    (short)0, acc1, false, false);
    }

    // Build next tile while WMMAs drain (tanh is a trans op -> co-executes
    // with the XDL pipe; global loads hide under WMMA latency). Uniform branch.
    if (s + 1 < kS) build_tile(s + 1, nxt);

    // Online softmax over s + img-weighted numerator.
    // Single-exp update: exactly one of {exp(m_old-m_new), exp(f-m_new)} is 1.
    const float* __restrict__ imgrow = imgb + (size_t)s * kD;
    const float ie0 = imgrow[e0], ie1 = imgrow[e0 + 16];
    #pragma unroll
    for (int v = 0; v < 8; ++v) {
      {
        float f = acc0[v] + bias0;
        float d = f - m0[v];
        float t = __expf(-fabsf(d));          // v_exp_f32
        bool  g = d > 0.f;
        float sc = g ? t : 1.f;
        float p  = g ? 1.f : t;
        l0[v] = fmaf(l0[v], sc, p);
        n0[v] = fmaf(n0[v], sc, ie0 * p);
        m0[v] = g ? f : m0[v];
      }
      {
        float f = acc1[v] + bias1;
        float d = f - m1[v];
        float t = __expf(-fabsf(d));
        bool  g = d > 0.f;
        float sc = g ? t : 1.f;
        float p  = g ? 1.f : t;
        l1[v] = fmaf(l1[v], sc, p);
        n1[v] = fmaf(n1[v], sc, ie1 * p);
        m1[v] = g ? f : m1[v];
      }
    }
  }

  // D layout: row c = v + 8*lane_half, col e = lane&15 (+tile offset)
  #pragma unroll
  for (int v = 0; v < 8; ++v) {
    int c = cbase + v + lhal * 8;
    if (c < kC) {
      out[((size_t)b * kC + c) * kD + e0]      = n0[v] / l0[v];
      out[((size_t)b * kC + c) * kD + e0 + 16] = n1[v] / l1[v];
    }
  }
}

extern "C" void kernel_launch(void* const* d_in, const int* in_sizes, int n_in,
                              void* d_out, int out_size, void* d_ws, size_t ws_size,
                              hipStream_t stream) {
  const float* img  = (const float*)d_in[0];
  const float* word = (const float*)d_in[1];
  const float* fw   = (const float*)d_in[2];
  const float* fb   = (const float*)d_in[3];
  float* out = (float*)d_out;

  const size_t shmem = kLdsHalfs * sizeof(_Float16);  // 294912 B (<320KB WGP LDS)
  hipFuncSetAttribute((const void*)&semdec_wmma_kernel,
                      hipFuncAttributeMaxDynamicSharedMemorySize, (int)shmem);

  dim3 grid(kCTiles, /*e halves*/ 2, kBS);  // 63 x 2 x 2 = 252 blocks
  semdec_wmma_kernel<<<grid, dim3(kThreads), shmem, stream>>>(img, word, fw, fb, out);
}
